// CausalMultiHeadAttention_45148696215585
// MI455X (gfx1250) — compile-verified, matrix operand order checked by reading, not run on hardware
//
#include <hip/hip_runtime.h>

// ---------------------------------------------------------------------------
// CDNA5 (gfx1250, wave32) causal multi-head attention
//   B=4, S=2048, D_MODEL=1024, H=16, DK=64
//
// Roofline: ~137 GFLOP total; working set (weights 8 MB f16, x 16 MB f16,
// Q/K/V/attn 64 MB f16) fits in the 192 MB L2 -> compute bound. All matmuls
// run on v_wmma_f32_16x16x32_f16 (K=32 path, 8x the per-instruction work of
// the f32 K=4 WMMA). GEMMs are register-blocked 64x64 per wave:
// 16 WMMA per 16 b128 loads per K-step. fp32->f16 conversion is hoisted into
// a one-time bandwidth-bound pass so inner loops have zero cvt VALU overhead.
// ---------------------------------------------------------------------------

#define DMODEL 1024
#define SLEN   2048
#define NHEAD  16
#define DK     64
#define NB     4

typedef __attribute__((ext_vector_type(16))) _Float16 v16h;
typedef __attribute__((ext_vector_type(8)))  _Float16 v8h;
typedef __attribute__((ext_vector_type(4)))  _Float16 v4h;
typedef __attribute__((ext_vector_type(4)))  float    v4f;
typedef __attribute__((ext_vector_type(8)))  float    v8f;

// D = A(16x32 f16) * B(32x16 f16) + C(16x16 f32)
__device__ __forceinline__ v8f wmma16(v16h a, v16h b, v8f c) {
  return __builtin_amdgcn_wmma_f32_16x16x32_f16(
      /*neg_a=*/false, a, /*neg_b=*/false, b,
      /*c_mod=*/(short)0, c, /*reuse_a=*/false, /*reuse_b=*/false);
}

// f16 fragment loader (global or LDS), 2x 16-byte loads.
// A 16x32 (ISA 7.12.2): lane lo = row, halves[0..7] = K hi*8+j,
// halves[8..15] = K 16+hi*8+j. B 32x16 mirrors this column-per-lane, so the
// same pattern applies with p = row of W (== column of B).
__device__ __forceinline__ v16h frag_f16(const _Float16* __restrict__ p) {
  v8h a = *(const v8h*)(p);
  v8h b = *(const v8h*)(p + 16);
  v16h f;
#pragma unroll
  for (int j = 0; j < 8; ++j) { f[j] = a[j]; f[j + 8] = b[j]; }
  return f;
}

// ---------------------------------------------------------------------------
// Kernel 0: one-time fp32 -> f16 conversion (pure bandwidth pass).
// ---------------------------------------------------------------------------
__global__ void __launch_bounds__(256)
cvt_f32_to_f16(const float* __restrict__ in, _Float16* __restrict__ out,
               int n4) {
  const int i = blockIdx.x * 256 + threadIdx.x;
  if (i < n4) {
    v4f v = ((const v4f*)in)[i];
    v4h h;
#pragma unroll
    for (int j = 0; j < 4; ++j) h[j] = (_Float16)v[j];
    ((v4h*)out)[i] = h;
  }
}

// ---------------------------------------------------------------------------
// Kernel 1: QKV projection, one wave per 64x64 output tile, blockIdx.z picks
// the matrix (0=Q, 1=K, 2=V).  out[m,n] = sum_k x[m,k]*W[n,k] + b[n].
// Per K-step: 4 A-frags + 4 B-frags (16 b128 loads) feed 16 WMMA.
// Q,K stored [token, h*64+d]; V stored transposed [b,h,dk,S] so the
// attention P*V B-fragments become contiguous loads.
// ---------------------------------------------------------------------------
__global__ void __launch_bounds__(32)
qkv_gemm_kernel(const _Float16* __restrict__ Xh,
                const _Float16* __restrict__ Wqh,
                const _Float16* __restrict__ Wkh,
                const _Float16* __restrict__ Wvh,
                const float* __restrict__ bq, const float* __restrict__ bk,
                const float* __restrict__ bv,
                _Float16* __restrict__ Qo, _Float16* __restrict__ Ko,
                _Float16* __restrict__ Vo) {
  const int lane = threadIdx.x;
  const int lo = lane & 15, hi = lane >> 4;
  const int n0 = blockIdx.x * 64;
  const int m0 = blockIdx.y * 64;
  const int z  = blockIdx.z;
  const _Float16* W    = (z == 0) ? Wqh : (z == 1) ? Wkh : Wvh;
  const float*    bias = (z == 0) ? bq  : (z == 1) ? bk  : bv;

  const _Float16* ar[4];
  const _Float16* br[4];
#pragma unroll
  for (int i = 0; i < 4; ++i) {
    ar[i] = Xh + (size_t)(m0 + i * 16 + lo) * DMODEL;
    br[i] = W  + (size_t)(n0 + i * 16 + lo) * DMODEL;
  }

  v8f acc[4][4] = {};
  for (int k = 0; k < DMODEL; k += 32) {
    v16h a[4];
#pragma unroll
    for (int i = 0; i < 4; ++i) a[i] = frag_f16(ar[i] + k + hi * 8);
#pragma unroll
    for (int j = 0; j < 4; ++j) {
      v16h bf = frag_f16(br[j] + k + hi * 8);
#pragma unroll
      for (int i = 0; i < 4; ++i) acc[i][j] = wmma16(a[i], bf, acc[i][j]);
    }
  }

  if (z < 2) {  // Q or K: row-major [token, feature]
    _Float16* O = (z == 0) ? Qo : Ko;
#pragma unroll
    for (int j = 0; j < 4; ++j) {
      const int col = n0 + j * 16 + lo;
      const float bb = bias[col];
#pragma unroll
      for (int i = 0; i < 4; ++i) {
#pragma unroll
        for (int r = 0; r < 8; ++r) {
          const int row = m0 + i * 16 + hi * 8 + r;  // C/D: VGPR r -> row
          O[(size_t)row * DMODEL + col] = (_Float16)(acc[i][j][r] + bb);
        }
      }
    }
  } else {      // V: transposed per head [b, h, dk, S] (contiguous in S)
    const int b_ = m0 >> 11;  // 64-row tile never crosses a batch boundary
#pragma unroll
    for (int j = 0; j < 4; ++j) {
      const int col = n0 + j * 16 + lo;
      const float bb = bias[col];
      const int hh = col >> 6, dl = col & 63;
      _Float16* vb = Vo + (((size_t)(b_ * NHEAD + hh) * DK) + dl) * SLEN;
#pragma unroll
      for (int i = 0; i < 4; ++i) {
        const int s0 = (m0 & (SLEN - 1)) + i * 16 + hi * 8;
#pragma unroll
        for (int r = 0; r < 8; ++r)
          vb[s0 + r] = (_Float16)(acc[i][j][r] + bb);
      }
    }
  }
}

// ---------------------------------------------------------------------------
// Kernel 2: flash-style causal attention, one wave per (b, h, 16-query tile).
// Per 32-key chunk: 4 WMMA for Q*K^T, online softmax via __shfl_xor row
// reductions (C layout: row in VGPR index, key in lane), P relayout C->A
// through LDS, 4 WMMA for P*V. Next chunk's K/V prefetched (global_prefetch).
// ---------------------------------------------------------------------------
__global__ void __launch_bounds__(32)
attn_kernel(const _Float16* __restrict__ Q, const _Float16* __restrict__ K,
            const _Float16* __restrict__ V, _Float16* __restrict__ O) {
  __shared__ __align__(16) _Float16 Plds[16][32];

  const int lane = threadIdx.x;
  const int lo = lane & 15, hi = lane >> 4;
  const int q0 = blockIdx.x * 16;
  const int h  = blockIdx.y;
  const int b  = blockIdx.z;

  const _Float16* qp = Q + (size_t)b * SLEN * DMODEL + h * DK;
  const _Float16* kp = K + (size_t)b * SLEN * DMODEL + h * DK;
  const _Float16* vp = V + (size_t)(b * NHEAD + h) * DK * SLEN;

  const v16h qa0 = frag_f16(qp + (size_t)(q0 + lo) * DMODEL + hi * 8);
  const v16h qa1 = frag_f16(qp + (size_t)(q0 + lo) * DMODEL + 32 + hi * 8);

  float mrow[8], lrow[8];
  v8f acc[4] = {{}, {}, {}, {}};
#pragma unroll
  for (int r = 0; r < 8; ++r) { mrow[r] = -__builtin_inff(); lrow[r] = 0.f; }

  const int s_end = q0 + 16;  // causal upper bound
  for (int s0 = 0; s0 < s_end; s0 += 32) {
    // Prefetch next chunk's K rows / V columns into cache (global_prefetch).
    __builtin_prefetch(kp + (size_t)(s0 + 32 + lo) * DMODEL, 0, 1);
    __builtin_prefetch(kp + (size_t)(s0 + 48 + lo) * DMODEL, 0, 1);
    __builtin_prefetch(vp + (size_t)lo * SLEN + s0 + 32, 0, 1);
    __builtin_prefetch(vp + (size_t)(48 + lo) * SLEN + s0 + 32, 0, 1);

    // ---- scores: S = Q * K^T ----
    v8f sc0 = {}, sc1 = {};
    {
      const _Float16* kr0 = kp + (size_t)(s0 + lo) * DMODEL;
      sc0 = wmma16(qa0, frag_f16(kr0 + hi * 8), sc0);
      sc0 = wmma16(qa1, frag_f16(kr0 + 32 + hi * 8), sc0);
      const _Float16* kr1 = kp + (size_t)(s0 + 16 + lo) * DMODEL;
      sc1 = wmma16(qa0, frag_f16(kr1 + hi * 8), sc1);
      sc1 = wmma16(qa1, frag_f16(kr1 + 32 + hi * 8), sc1);
    }

    // ---- scale + causal mask ----
    const int key0 = s0 + lo, key1 = s0 + 16 + lo;
#pragma unroll
    for (int r = 0; r < 8; ++r) {
      const int qrow = q0 + hi * 8 + r;
      const float a0 = sc0[r] * 0.125f;   // 1/sqrt(64)
      const float a1 = sc1[r] * 0.125f;
      sc0[r] = (key0 <= qrow) ? a0 : -__builtin_inff();
      sc1[r] = (key1 <= qrow) ? a1 : -__builtin_inff();
    }

    // ---- online softmax (row spread across 16 lanes of a half-wave) ----
#pragma unroll
    for (int r = 0; r < 8; ++r) {
      float cm = fmaxf(sc0[r], sc1[r]);
      cm = fmaxf(cm, __shfl_xor(cm, 1));
      cm = fmaxf(cm, __shfl_xor(cm, 2));
      cm = fmaxf(cm, __shfl_xor(cm, 4));
      cm = fmaxf(cm, __shfl_xor(cm, 8));
      const float mnew = fmaxf(mrow[r], cm);
      const float corr = __expf(mrow[r] - mnew);
      const float p0 = __expf(sc0[r] - mnew);
      const float p1 = __expf(sc1[r] - mnew);
      float rs = p0 + p1;
      rs += __shfl_xor(rs, 1);
      rs += __shfl_xor(rs, 2);
      rs += __shfl_xor(rs, 4);
      rs += __shfl_xor(rs, 8);
      lrow[r] = lrow[r] * corr + rs;
      mrow[r] = mnew;
#pragma unroll
      for (int t = 0; t < 4; ++t) acc[t][r] *= corr;
      sc0[r] = p0;
      sc1[r] = p1;
    }

    // ---- relayout P: C-fragment -> A-fragment via LDS ----
    __syncthreads();
#pragma unroll
    for (int r = 0; r < 8; ++r) {
      Plds[hi * 8 + r][lo]      = (_Float16)sc0[r];
      Plds[hi * 8 + r][16 + lo] = (_Float16)sc1[r];
    }
    __syncthreads();
    const v16h pa = frag_f16(&Plds[lo][hi * 8]);

    // ---- out += P * V (V transposed: column d contiguous in S) ----
#pragma unroll
    for (int t = 0; t < 4; ++t) {
      const _Float16* vc = vp + (size_t)(t * 16 + lo) * SLEN + s0;
      acc[t] = wmma16(pa, frag_f16(vc + hi * 8), acc[t]);
    }
  }

  // ---- finalize: divide by row sum, store [b, s, h*64+d] (f16) ----
#pragma unroll
  for (int r = 0; r < 8; ++r) {
    const int qrow = q0 + hi * 8 + r;
    const float inv = 1.0f / lrow[r];
    _Float16* orow = O + ((size_t)b * SLEN + qrow) * DMODEL + h * DK;
#pragma unroll
    for (int t = 0; t < 4; ++t)
      orow[t * 16 + lo] = (_Float16)(acc[t][r] * inv);
  }
}

// ---------------------------------------------------------------------------
// Kernel 3: output projection, 64x64 tile per wave. out = attn @ Wo.T + bo,
// fp32 result straight to d_out.
// ---------------------------------------------------------------------------
__global__ void __launch_bounds__(32)
out_proj_kernel(const _Float16* __restrict__ A, const _Float16* __restrict__ W,
                const float* __restrict__ bias, float* __restrict__ out) {
  const int lane = threadIdx.x;
  const int lo = lane & 15, hi = lane >> 4;
  const int n0 = blockIdx.x * 64;
  const int m0 = blockIdx.y * 64;

  const _Float16* ar[4];
  const _Float16* br[4];
#pragma unroll
  for (int i = 0; i < 4; ++i) {
    ar[i] = A + (size_t)(m0 + i * 16 + lo) * DMODEL;
    br[i] = W + (size_t)(n0 + i * 16 + lo) * DMODEL;
  }

  v8f acc[4][4] = {};
  for (int k = 0; k < DMODEL; k += 32) {
    v16h a[4];
#pragma unroll
    for (int i = 0; i < 4; ++i) a[i] = frag_f16(ar[i] + k + hi * 8);
#pragma unroll
    for (int j = 0; j < 4; ++j) {
      v16h bf = frag_f16(br[j] + k + hi * 8);
#pragma unroll
      for (int i = 0; i < 4; ++i) acc[i][j] = wmma16(a[i], bf, acc[i][j]);
    }
  }

#pragma unroll
  for (int j = 0; j < 4; ++j) {
    const int col = n0 + j * 16 + lo;
    const float bb = bias[col];
#pragma unroll
    for (int i = 0; i < 4; ++i) {
#pragma unroll
      for (int r = 0; r < 8; ++r) {
        const int row = m0 + i * 16 + hi * 8 + r;
        out[(size_t)row * DMODEL + col] = acc[i][j][r] + bb;
      }
    }
  }
}

// ---------------------------------------------------------------------------
extern "C" void kernel_launch(void* const* d_in, const int* in_sizes, int n_in,
                              void* d_out, int out_size, void* d_ws,
                              size_t ws_size, hipStream_t stream) {
  const float* x  = (const float*)d_in[0];
  const float* Wq = (const float*)d_in[1];
  const float* bq = (const float*)d_in[2];
  const float* Wk = (const float*)d_in[3];
  const float* bk = (const float*)d_in[4];
  const float* Wv = (const float*)d_in[5];
  const float* bv = (const float*)d_in[6];
  const float* Wo = (const float*)d_in[7];
  const float* bo = (const float*)d_in[8];
  float* out = (float*)d_out;

  const size_t TOK = (size_t)NB * SLEN;          // 8192 tokens
  const size_t MAT = TOK * DMODEL;               // 8M elements
  const size_t WEL = (size_t)DMODEL * DMODEL;    // 1M elements

  // Workspace map (f16): q/k/v/attn 16 MB each, x_h 16 MB, weights 2 MB x4.
  char* ws = (char*)d_ws;
  _Float16* q_ws = (_Float16*)(ws);
  _Float16* k_ws = (_Float16*)(ws + 16ull * 1024 * 1024);
  _Float16* v_ws = (_Float16*)(ws + 32ull * 1024 * 1024);
  _Float16* a_ws = (_Float16*)(ws + 48ull * 1024 * 1024);
  _Float16* x_h  = (_Float16*)(ws + 64ull * 1024 * 1024);
  _Float16* wq_h = (_Float16*)(ws + 80ull * 1024 * 1024);
  _Float16* wk_h = (_Float16*)(ws + 82ull * 1024 * 1024);
  _Float16* wv_h = (_Float16*)(ws + 84ull * 1024 * 1024);
  _Float16* wo_h = (_Float16*)(ws + 86ull * 1024 * 1024);

  // 0) fp32 -> f16 conversion passes (bandwidth-bound, L2-resident after).
  cvt_f32_to_f16<<<(int)(MAT / 4 / 256), 256, 0, stream>>>(x, x_h, (int)(MAT / 4));
  cvt_f32_to_f16<<<(int)(WEL / 4 / 256), 256, 0, stream>>>(Wq, wq_h, (int)(WEL / 4));
  cvt_f32_to_f16<<<(int)(WEL / 4 / 256), 256, 0, stream>>>(Wk, wk_h, (int)(WEL / 4));
  cvt_f32_to_f16<<<(int)(WEL / 4 / 256), 256, 0, stream>>>(Wv, wv_h, (int)(WEL / 4));
  cvt_f32_to_f16<<<(int)(WEL / 4 / 256), 256, 0, stream>>>(Wo, wo_h, (int)(WEL / 4));

  // 1) QKV projections: 64x64 tile per wave, z selects Q/K/V.
  dim3 qkv_grid(DMODEL / 64, (unsigned)(TOK / 64), 3);
  qkv_gemm_kernel<<<qkv_grid, 32, 0, stream>>>(x_h, wq_h, wk_h, wv_h,
                                               bq, bk, bv, q_ws, k_ws, v_ws);

  // 2) Attention: one wave per (query tile, head, batch).
  dim3 attn_grid(SLEN / 16, NHEAD, NB);
  attn_kernel<<<attn_grid, 32, 0, stream>>>(q_ws, k_ws, v_ws, a_ws);

  // 3) Output projection (fp32 result).
  dim3 proj_grid(DMODEL / 64, (unsigned)(TOK / 64));
  out_proj_kernel<<<proj_grid, 32, 0, stream>>>(a_ws, wo_h, bo, out);
}
